// ACNN_41781441855613
// MI455X (gfx1250) — matrix operand order repaired and checked by reading, not compiled, not run on hardware
//
#include <hip/hip_runtime.h>

#define Bb 16
#define Ss 2048
#define Ff 512
#define Hh 512
#define Wn 5
#define Mm (Bb * Ss)      // 32768 rows

typedef __attribute__((ext_vector_type(8)))  float    v8f;
typedef __attribute__((ext_vector_type(16))) _Float16 v16h;
typedef __attribute__((ext_vector_type(8)))  _Float16 v8h;

union V16 { v16h v; v8h h[2]; };

#define BM 128
#define BN 128
#define BK 32
#define LDT (BK + 8)      // padded LDS row stride (halves); keeps 16B alignment

// ---------------- CDNA5 async copy helpers ----------------
__device__ __forceinline__ unsigned lds_off(const void* p) {
    // generic pointer to LDS: low 32 bits are the workgroup-relative LDS byte address
    return (unsigned)(uintptr_t)p;
}
// GVS mode: mem_addr = SGPR_base + VGPR_i32_offset ; LDS dest from VGPR. Tracked by ASYNCcnt.
__device__ __forceinline__ void async_b128_s(unsigned lds_addr, int voff_bytes,
                                             unsigned long long sbase) {
    asm volatile("global_load_async_to_lds_b128 %0, %1, %2"
                 :: "v"(lds_addr), "v"(voff_bytes), "s"(sbase)
                 : "memory");
}
__device__ __forceinline__ void wait_async0() {
    asm volatile("s_wait_asynccnt 0" ::: "memory");
}

// ---------------- prep kernels ----------------
__global__ void k_f32_to_f16(const float* __restrict__ src, _Float16* __restrict__ dst, int n) {
    int i = blockIdx.x * blockDim.x + threadIdx.x;
    if (i < n) dst[i] = (_Float16)src[i];
}

// src [rows][cols] f32 (row-major) -> dst [cols][rows] f16
__global__ void k_transpose_f16(const float* __restrict__ src, _Float16* __restrict__ dst,
                                int rows, int cols) {
    int i = blockIdx.x * blockDim.x + threadIdx.x;
    if (i >= rows * cols) return;
    int r = i / cols, c = i - r * cols;
    dst[(size_t)c * rows + r] = (_Float16)src[i];
}

// ---------------- GEMM 1: QK[M,1024] = Xh[M,F] x [Wq|Wk] ----------------
__global__ __launch_bounds__(256)
void k_gemm_qk(const _Float16* __restrict__ Xh, const _Float16* __restrict__ WT,
               float* __restrict__ C) {
    __shared__ _Float16 lA[2][BM * LDT];
    __shared__ _Float16 lB[2][BN * LDT];
    const int Kdim = Ff;          // 512
    const int N    = 2 * Hh;      // 1024
    const int t = threadIdx.x;
    const int mBase = blockIdx.y * BM;
    const int nBase = blockIdx.x * BN;
    const int wave = t >> 5, lane = t & 31;
    const int wm  = (wave & 3) * 32;      // 4 waves along M
    const int wnn = (wave >> 2) * 64;     // 2 waves along N, 64 cols each
    const int laneN = lane & 15, laneSel = lane >> 4;
    const int r0 = t >> 2, ch = t & 3;    // 128 rows x 4 chunks of 8 halves

    // loop-invariant per-lane byte offsets (i32) and uniform advancing bases (SGPR)
    int aOff[2], bOff[2];
#pragma unroll
    for (int i = 0; i < 2; ++i) {
        int row = r0 + i * 64;
        aOff[i] = ((mBase + row) * Ff + ch * 8) * 2;
        bOff[i] = ((nBase + row) * Kdim + ch * 8) * 2;
    }
    const unsigned long long aBase = (unsigned long long)(uintptr_t)Xh;
    const unsigned long long bBase = (unsigned long long)(uintptr_t)WT;

    v8f acc[2][4] = {};

    // prologue: async-fill buffer 0  (k0 = 0)
#pragma unroll
    for (int i = 0; i < 2; ++i) {
        int row = r0 + i * 64;
        async_b128_s(lds_off(&lA[0][row * LDT + ch * 8]), aOff[i], aBase);
        async_b128_s(lds_off(&lB[0][row * LDT + ch * 8]), bOff[i], bBase);
    }
    int buf = 0;
    const int nk = Kdim / BK;     // 16
    for (int ks = 0; ks < nk; ++ks) {
        wait_async0();
        __syncthreads();
        if (ks + 1 < nk) {
            unsigned long long kb = (unsigned long long)((ks + 1) * BK * 2);
#pragma unroll
            for (int i = 0; i < 2; ++i) {
                int row = r0 + i * 64;
                async_b128_s(lds_off(&lA[buf ^ 1][row * LDT + ch * 8]), aOff[i], aBase + kb);
                async_b128_s(lds_off(&lB[buf ^ 1][row * LDT + ch * 8]), bOff[i], bBase + kb);
            }
        }
        V16 a[2], b[4];
#pragma unroll
        for (int i = 0; i < 2; ++i) {
            const _Float16* p = &lA[buf][(wm + i * 16 + laneN) * LDT + laneSel * 8];
            a[i].h[0] = *(const v8h*)p;
            a[i].h[1] = *(const v8h*)(p + 16);
        }
#pragma unroll
        for (int j = 0; j < 4; ++j) {
            const _Float16* p = &lB[buf][(wnn + j * 16 + laneN) * LDT + laneSel * 8];
            b[j].h[0] = *(const v8h*)p;
            b[j].h[1] = *(const v8h*)(p + 16);
        }
#pragma unroll
        for (int i = 0; i < 2; ++i)
#pragma unroll
            for (int j = 0; j < 4; ++j)
                acc[i][j] = __builtin_amdgcn_wmma_f32_16x16x32_f16(
                    false, a[i].v, false, b[j].v, (short)0, acc[i][j], false, false);
        buf ^= 1;
    }
#pragma unroll
    for (int i = 0; i < 2; ++i)
#pragma unroll
        for (int j = 0; j < 4; ++j) {
            int rbase = mBase + wm + i * 16 + laneSel * 8;
            int cidx  = nBase + wnn + j * 16 + laneN;
#pragma unroll
            for (int e = 0; e < 8; ++e)
                C[(size_t)(rbase + e) * N + cidx] = acc[i][j][e];
        }
}

// ---------------- scores + softmax: one wave per position ----------------
__global__ __launch_bounds__(256)
void k_scores(const float* __restrict__ QK, const float* __restrict__ v_att,
              float* __restrict__ att) {
    int m = blockIdx.x * 8 + (threadIdx.x >> 5);
    int lane = threadIdx.x & 31;
    const float* Qrow = QK + (size_t)m * 1024;
    float q[16], vv[16];
#pragma unroll
    for (int j = 0; j < 16; ++j) {
        int h = lane + 32 * j;
        q[j] = Qrow[h];
        vv[j] = v_att[h];
    }
    int s = m & (Ss - 1);
    float sc[Wn];
#pragma unroll
    for (int w = 0; w < Wn; ++w) {
        int sr = s + w - 2;
        bool ok = (unsigned)sr < (unsigned)Ss;
        const float* Krow = QK + (long)(m + w - 2) * 1024 + 512;
        float a = 0.f;
#pragma unroll
        for (int j = 0; j < 16; ++j) {
            float kv = ok ? Krow[lane + 32 * j] : 0.f;   // padded row -> kproj == 0
            a += tanhf(kv + q[j]) * vv[j];
        }
#pragma unroll
        for (int off = 16; off; off >>= 1) a += __shfl_xor(a, off, 32);
        sc[w] = a;
    }
    float mx = sc[0];
#pragma unroll
    for (int w = 1; w < Wn; ++w) mx = fmaxf(mx, sc[w]);
    float den = 0.f;
#pragma unroll
    for (int w = 0; w < Wn; ++w) { sc[w] = __expf(sc[w] - mx); den += sc[w]; }
    float inv = 1.0f / den;
    if (lane < Wn) att[m * Wn + lane] = sc[lane] * inv;
}

// ---------------- GEMM 2: out = [att-weighted windows][M,2560] x cnn_W + b ----------------
__global__ __launch_bounds__(256)
void k_gemm_ctx(const _Float16* __restrict__ Xh, const _Float16* __restrict__ CwT,
                const float* __restrict__ att, const float* __restrict__ bias,
                float* __restrict__ out) {
    __shared__ _Float16 lA[2][BM * LDT];
    __shared__ _Float16 lB[2][BN * LDT];
    const int Kdim = Wn * Ff;     // 2560
    const int t = threadIdx.x;
    const int mBase = blockIdx.y * BM;
    const int nBase = blockIdx.x * BN;
    const int wave = t >> 5, lane = t & 31;
    const int wm  = (wave & 3) * 32;
    const int wnn = (wave >> 2) * 64;
    const int laneN = lane & 15, laneSel = lane >> 4;
    const int r0 = t >> 2, ch = t & 3;

    // per-thread row state, hoisted out of the k-loop
    const int mrow[2] = { mBase + r0, mBase + r0 + 64 };
    const int srow[2] = { mrow[0] & (Ss - 1), mrow[1] & (Ss - 1) };
    float attv[2][Wn];
#pragma unroll
    for (int i = 0; i < 2; ++i)
#pragma unroll
        for (int w = 0; w < Wn; ++w) attv[i][w] = att[mrow[i] * Wn + w];

    int bOff[2];
#pragma unroll
    for (int i = 0; i < 2; ++i)
        bOff[i] = ((nBase + r0 + i * 64) * Kdim + ch * 8) * 2;
    const unsigned long long bBase = (unsigned long long)(uintptr_t)CwT;

    v8h areg[2];
    auto loadA = [&](int k0) {                 // synthesize att-scaled shifted A chunks
        int w = k0 >> 9;                       // window slot; (m+w-2)*F + f0 == (m-2)*F + k0
#pragma unroll
        for (int i = 0; i < 2; ++i) {
            int sr = srow[i] + w - 2;
            v8h val = {};
            if ((unsigned)sr < (unsigned)Ss) {
                val = *(const v8h*)(Xh + (long)(mrow[i] - 2) * Ff + k0 + ch * 8);
                float sc = attv[i][w];
#pragma unroll
                for (int e = 0; e < 8; ++e) val[e] = (_Float16)((float)val[e] * sc);
            }
            areg[i] = val;
        }
    };
    auto storeA = [&](int b) {
#pragma unroll
        for (int i = 0; i < 2; ++i)
            *(v8h*)&lA[b][(r0 + i * 64) * LDT + ch * 8] = areg[i];
    };
    auto issueB = [&](int b, int k0) {
        unsigned long long kb = (unsigned long long)(k0 * 2);
#pragma unroll
        for (int i = 0; i < 2; ++i)
            async_b128_s(lds_off(&lB[b][(r0 + i * 64) * LDT + ch * 8]), bOff[i], bBase + kb);
    };

    v8f acc[2][4] = {};
    loadA(0); storeA(0); issueB(0, 0);
    int buf = 0;
    const int nk = Kdim / BK;     // 80
    for (int ks = 0; ks < nk; ++ks) {
        wait_async0();
        __syncthreads();
        if (ks + 1 < nk) { issueB(buf ^ 1, (ks + 1) * BK); loadA((ks + 1) * BK); }
        V16 a[2], b[4];
#pragma unroll
        for (int i = 0; i < 2; ++i) {
            const _Float16* p = &lA[buf][(wm + i * 16 + laneN) * LDT + laneSel * 8];
            a[i].h[0] = *(const v8h*)p;
            a[i].h[1] = *(const v8h*)(p + 16);
        }
#pragma unroll
        for (int j = 0; j < 4; ++j) {
            const _Float16* p = &lB[buf][(wnn + j * 16 + laneN) * LDT + laneSel * 8];
            b[j].h[0] = *(const v8h*)p;
            b[j].h[1] = *(const v8h*)(p + 16);
        }
#pragma unroll
        for (int i = 0; i < 2; ++i)
#pragma unroll
            for (int j = 0; j < 4; ++j)
                acc[i][j] = __builtin_amdgcn_wmma_f32_16x16x32_f16(
                    false, a[i].v, false, b[j].v, (short)0, acc[i][j], false, false);
        if (ks + 1 < nk) storeA(buf ^ 1);
        buf ^= 1;
    }
#pragma unroll
    for (int i = 0; i < 2; ++i)
#pragma unroll
        for (int j = 0; j < 4; ++j) {
            int rbase = mBase + wm + i * 16 + laneSel * 8;
            int cidx  = nBase + wnn + j * 16 + laneN;
            float bv = bias[cidx];
#pragma unroll
            for (int e = 0; e < 8; ++e)
                out[(size_t)(rbase + e) * Hh + cidx] = acc[i][j][e] + bv;
        }
}

// ---------------- host ----------------
extern "C" void kernel_launch(void* const* d_in, const int* in_sizes, int n_in,
                              void* d_out, int out_size, void* d_ws, size_t ws_size,
                              hipStream_t stream) {
    (void)in_sizes; (void)n_in; (void)out_size; (void)ws_size;
    const float* emb   = (const float*)d_in[0];
    const float* Wq    = (const float*)d_in[1];
    const float* Wk    = (const float*)d_in[2];
    const float* v_att = (const float*)d_in[3];
    const float* cnn_W = (const float*)d_in[4];
    const float* cnn_b = (const float*)d_in[5];
    float* out = (float*)d_out;

    char* ws = (char*)d_ws;                           // 256B-aligned regions
    _Float16* Xh   = (_Float16*)(ws);                 // [M][F]    f16   33.5 MB
    _Float16* WqkT = (_Float16*)(ws + 33554432);      // [2H][F]   f16    1.0 MB
    _Float16* CwT  = (_Float16*)(ws + 34603008);      // [H][5F]   f16    2.6 MB
    float*    QK   = (float*)   (ws + 37224448);      // [M][2H]   f32  134.2 MB
    float*    att  = (float*)   (ws + 171442176);     // [M][W]    f32    0.7 MB

    k_f32_to_f16<<<(Mm * Ff + 255) / 256, 256, 0, stream>>>(emb, Xh, Mm * Ff);
    k_transpose_f16<<<(Ff * Hh + 255) / 256, 256, 0, stream>>>(Wq, WqkT, Ff, Hh);
    k_transpose_f16<<<(Ff * Hh + 255) / 256, 256, 0, stream>>>(Wk, WqkT + (size_t)Hh * Ff, Ff, Hh);
    k_transpose_f16<<<(Wn * Ff * Hh + 255) / 256, 256, 0, stream>>>(cnn_W, CwT, Wn * Ff, Hh);

    k_gemm_qk<<<dim3((2 * Hh) / BN, Mm / BM), 256, 0, stream>>>(Xh, WqkT, QK);
    k_scores<<<Mm / 8, 256, 0, stream>>>(QK, v_att, att);
    k_gemm_ctx<<<dim3(Hh / BN, Mm / BM), 256, 0, stream>>>(Xh, CwT, att, cnn_b, out);
}
